// MatMulFreeGLU_21706764714142
// MI455X (gfx1250) — 8-bit matrix operand order repaired and checked by reading, compiled, not run on hardware
//
#include <hip/hip_runtime.h>
#include <cstdint>


typedef int v8i __attribute__((ext_vector_type(8)));

constexpr int TOK = 8192;
constexpr int IND = 2048;
constexpr int HID = 4096;

constexpr size_t OFF_XQ   = 0;
constexpr size_t OFF_INV  = OFF_XQ   + (size_t)TOK * IND;
constexpr size_t OFF_WG   = OFF_INV  + (size_t)TOK * 4;
constexpr size_t OFF_WU   = OFF_WG   + (size_t)HID * IND;
constexpr size_t OFF_WD   = OFF_WU   + (size_t)HID * IND;
constexpr size_t OFF_P    = OFF_WD   + (size_t)HID * HID;
constexpr size_t OFF_PL0  = OFF_P    + (size_t)TOK * HID * 4;
constexpr size_t OFF_PL1  = OFF_PL0  + (size_t)TOK * HID;
constexpr size_t OFF_PL2  = OFF_PL1  + (size_t)TOK * HID;
constexpr size_t OFF_PART = OFF_PL2  + (size_t)TOK * HID;
constexpr size_t OFF_S    = OFF_PART + 2048 * 4;

__device__ __forceinline__ float sigmoidf(float x) { return 1.0f / (1.0f + __expf(-x)); }

__device__ __forceinline__ v8i zero_v8i() {
    v8i z;
#pragma unroll
    for (int i = 0; i < 8; ++i) z[i] = 0;
    return z;
}

__device__ __forceinline__ v8i load_a64o(const int8_t* __restrict__ base, int off) {
    const int2 t0 = *(const int2*)(base + off);
    const int2 t1 = *(const int2*)(base + off + 8);
    const int2 t2 = *(const int2*)(base + off + 32);
    const int2 t3 = *(const int2*)(base + off + 40);
    v8i a;
    a[0] = t0.x; a[1] = t0.y; a[2] = t1.x; a[3] = t1.y;
    a[4] = t2.x; a[5] = t2.y; a[6] = t3.x; a[7] = t3.y;
    return a;
}

__device__ __forceinline__ v8i load_b64o(const int8_t* __restrict__ base, int off) {
    const int4 u0 = *(const int4*)(base + off);
    const int4 u1 = *(const int4*)(base + off + 32);
    v8i b;
    b[0] = u0.x; b[1] = u0.y; b[2] = u0.z; b[3] = u0.w;
    b[4] = u1.x; b[5] = u1.y; b[6] = u1.z; b[7] = u1.w;
    return b;
}

#define WMMA_IU8(A, B, C) __builtin_amdgcn_wmma_i32_16x16x64_iu8(true, (A), true, (B), (C), false, false)

__global__ __launch_bounds__(256) void absmean_partial(const float* __restrict__ w, int n,
                                                       float* __restrict__ partial) {
    __shared__ float sh[256];
    float s = 0.0f;
    for (int i = blockIdx.x * 256 + threadIdx.x; i < n; i += gridDim.x * 256)
        s += fabsf(w[i]);
    sh[threadIdx.x] = s;
    __syncthreads();
    for (int k = 128; k > 0; k >>= 1) {
        if ((int)threadIdx.x < k) sh[threadIdx.x] += sh[threadIdx.x + k];
        __syncthreads();
    }
    if (threadIdx.x == 0) partial[blockIdx.x] = sh[0];
}

__global__ __launch_bounds__(256) void absmean_final(const float* __restrict__ partial, int np,
                                                     long n, float* __restrict__ s_out) {
    __shared__ double sh[256];
    double s = 0.0;
    for (int i = threadIdx.x; i < np; i += 256) s += (double)partial[i];
    sh[threadIdx.x] = s;
    __syncthreads();
    for (int k = 128; k > 0; k >>= 1) {
        if ((int)threadIdx.x < k) sh[threadIdx.x] += sh[threadIdx.x + k];
        __syncthreads();
    }
    if (threadIdx.x == 0) {
        float mean = (float)(sh[0] / (double)n);
        float sc = 1.0f / (mean + 1e-8f);
        sc = fminf(fmaxf(sc, 0.001f), 1000.0f);
        s_out[0] = sc;
    }
}

__global__ __launch_bounds__(256) void wmask_kernel(const float* __restrict__ w,
                                                    const float* __restrict__ sp,
                                                    int8_t* __restrict__ m, int n) {
    const float s = sp[0];
    for (int i = blockIdx.x * 256 + threadIdx.x; i < n; i += gridDim.x * 256) {
        float t  = fminf(fmaxf(rintf(s * w[i]), -1.0f), 1.0f);
        float wq = t / s;
        int8_t mv = (wq == 1.0f) ? (int8_t)1 : ((wq == -1.0f) ? (int8_t)-1 : (int8_t)0);
        m[i] = mv;
    }
}

__global__ __launch_bounds__(256) void rmsq_kernel(const float* __restrict__ x,
                                                   int8_t* __restrict__ xq,
                                                   float* __restrict__ inv_s) {
    __shared__ float sh[256];
    const int row = blockIdx.x;
    const int tid = threadIdx.x;
    const float* xr = x + (size_t)row * IND;

    float v[8];
#pragma unroll
    for (int i = 0; i < 8; ++i) v[i] = xr[tid + i * 256];

    float s = 0.0f, sq = 0.0f;
#pragma unroll
    for (int i = 0; i < 8; ++i) { s += v[i]; sq += v[i] * v[i]; }

    sh[tid] = s; __syncthreads();
    for (int k = 128; k > 0; k >>= 1) { if (tid < k) sh[tid] += sh[tid + k]; __syncthreads(); }
    const float mu = sh[0] * (1.0f / IND);
    __syncthreads();

    sh[tid] = sq; __syncthreads();
    for (int k = 128; k > 0; k >>= 1) { if (tid < k) sh[tid] += sh[tid + k]; __syncthreads(); }
    float var = fmaxf(sh[0] * (1.0f / IND) - mu * mu, 0.0f);
    __syncthreads();

    const float r = fminf(1.0f / sqrtf(var + 1e-8f), 1000.0f);

    float mx = 0.0f;
#pragma unroll
    for (int i = 0; i < 8; ++i) {
        v[i] = r * (v[i] - mu);
        mx = fmaxf(mx, fabsf(v[i]));
    }
    sh[tid] = mx; __syncthreads();
    for (int k = 128; k > 0; k >>= 1) { if (tid < k) sh[tid] = fmaxf(sh[tid], sh[tid + k]); __syncthreads(); }
    mx = sh[0];

    float sact = 127.0f / (mx + 1e-8f);
    sact = fminf(fmaxf(sact, 0.001f), 1000.0f);

    int8_t* qr = xq + (size_t)row * IND;
#pragma unroll
    for (int i = 0; i < 8; ++i) {
        float q = fminf(fmaxf(rintf(sact * v[i]), -128.0f), 127.0f);
        qr[tid + i * 256] = (int8_t)q;
    }
    if (tid == 0) inv_s[row] = 1.0f / sact;
}

__global__ __launch_bounds__(256) void gemm1_kernel(const int8_t* __restrict__ xq,
                                                    const float* __restrict__ inv_s,
                                                    const int8_t* __restrict__ wg,
                                                    const int8_t* __restrict__ wu,
                                                    float* __restrict__ p_out,
                                                    int8_t* __restrict__ pl0,
                                                    int8_t* __restrict__ pl1,
                                                    int8_t* __restrict__ pl2) {
    const int wave = threadIdx.x >> 5;
    const int lane = threadIdx.x & 31;
    const int half = lane >> 4;
    const int lr   = lane & 15;

    const int row0 = blockIdx.x * 128 + wave * 16;
    const int col0 = blockIdx.y * 32;
    constexpr int BT = 16 * IND;

    v8i accG[2], accU[2];
#pragma unroll
    for (int t = 0; t < 2; ++t) { accG[t] = zero_v8i(); accU[t] = zero_v8i(); }

    const int avoff = (row0 + lr) * IND + 16 * half;
    const int bvoff = (col0 + lr) * IND + 16 * half;

#pragma unroll
    for (int k = 0; k < IND; k += 64) {
        const v8i a = load_a64o(xq, avoff + k);
#pragma unroll
        for (int t = 0; t < 2; ++t) {
            const v8i bg = load_b64o(wg, bvoff + t * BT + k);
            accG[t] = WMMA_IU8(a, bg, accG[t]);
            const v8i bu = load_b64o(wu, bvoff + t * BT + k);
            accU[t] = WMMA_IU8(a, bu, accU[t]);
        }
    }

#pragma unroll
    for (int t = 0; t < 2; ++t) {
        const int col = col0 + t * 16 + lr;
#pragma unroll
        for (int r = 0; r < 8; ++r) {
            const int row = row0 + r + 8 * half;
            const float is = inv_s[row];
            const int gi = accG[t][r];
            const float g = (float)gi * is;
            const float u = (float)accU[t][r] * is;
            const size_t idx = (size_t)row * HID + col;
            p_out[idx] = sigmoidf(g) * u;

            const int b0 = ((gi + 128) & 255) - 128;
            const int r1 = (gi - b0) >> 8;
            const int b1 = ((r1 + 128) & 255) - 128;
            const int b2 = (r1 - b1) >> 8;
            pl0[idx] = (int8_t)b0;
            pl1[idx] = (int8_t)b1;
            pl2[idx] = (int8_t)b2;
        }
    }
}

__global__ __launch_bounds__(256) void gemm2_kernel(const int8_t* __restrict__ pl0,
                                                    const int8_t* __restrict__ pl1,
                                                    const int8_t* __restrict__ pl2,
                                                    const int8_t* __restrict__ wd,
                                                    const float* __restrict__ inv_s,
                                                    const float* __restrict__ p_in,
                                                    float* __restrict__ out) {
    const int wave = threadIdx.x >> 5;
    const int lane = threadIdx.x & 31;
    const int half = lane >> 4;
    const int lr   = lane & 15;

    const int row0 = blockIdx.x * 128 + wave * 16;
    const int col0 = blockIdx.y * 32;
    constexpr int BT = 16 * HID;

    v8i acc0[2], acc1[2], acc2[2];
#pragma unroll
    for (int t = 0; t < 2; ++t) { acc0[t] = zero_v8i(); acc1[t] = zero_v8i(); acc2[t] = zero_v8i(); }

    const int avoff = (row0 + lr) * HID + 16 * half;
    const int bvoff = (col0 + lr) * HID + 16 * half;

#pragma unroll
    for (int k = 0; k < HID; k += 64) {
        const v8i a0 = load_a64o(pl0, avoff + k);
        const v8i a1 = load_a64o(pl1, avoff + k);
        const v8i a2 = load_a64o(pl2, avoff + k);
#pragma unroll
        for (int t = 0; t < 2; ++t) {
            const v8i b = load_b64o(wd, bvoff + t * BT + k);
            acc0[t] = WMMA_IU8(a0, b, acc0[t]);
            acc1[t] = WMMA_IU8(a1, b, acc1[t]);
            acc2[t] = WMMA_IU8(a2, b, acc2[t]);
        }
    }

#pragma unroll
    for (int t = 0; t < 2; ++t) {
        const int col = col0 + t * 16 + lr;
#pragma unroll
        for (int r = 0; r < 8; ++r) {
            const int row = row0 + r + 8 * half;
            const float is = inv_s[row];
            const long di = (long)acc0[t][r] + 256L * (long)acc1[t][r] + 65536L * (long)acc2[t][r];
            const float d = (float)((double)di * (double)is);
            const size_t idx = (size_t)row * HID + col;
            out[idx] = sigmoidf(d) * p_in[idx];
        }
    }
}

extern "C" void kernel_launch(void* const* d_in, const int* in_sizes, int n_in,
                              void* d_out, int out_size, void* d_ws, size_t ws_size,
                              hipStream_t stream) {
    (void)in_sizes; (void)n_in; (void)out_size; (void)ws_size;

    const float* x  = (const float*)d_in[0];
    const float* Wg = (const float*)d_in[1];
    const float* Wu = (const float*)d_in[2];
    const float* Wd = (const float*)d_in[3];
    float* out = (float*)d_out;

    char* ws = (char*)d_ws;
    int8_t* xq   = (int8_t*)(ws + OFF_XQ);
    float*  invs = (float*)(ws + OFF_INV);
    int8_t* wgm  = (int8_t*)(ws + OFF_WG);
    int8_t* wum  = (int8_t*)(ws + OFF_WU);
    int8_t* wdm  = (int8_t*)(ws + OFF_WD);
    float*  p    = (float*)(ws + OFF_P);
    int8_t* pl0  = (int8_t*)(ws + OFF_PL0);
    int8_t* pl1  = (int8_t*)(ws + OFF_PL1);
    int8_t* pl2  = (int8_t*)(ws + OFF_PL2);
    float*  part = (float*)(ws + OFF_PART);
    float*  sbuf = (float*)(ws + OFF_S);

    const float* wptr[3] = { Wg, Wu, Wd };
    int8_t*      mptr[3] = { wgm, wum, wdm };
    const int    wn[3]   = { HID * IND, HID * IND, HID * HID };
    for (int i = 0; i < 3; ++i) {
        absmean_partial<<<2048, 256, 0, stream>>>(wptr[i], wn[i], part);
        absmean_final<<<1, 256, 0, stream>>>(part, 2048, (long)wn[i], sbuf + i);
        wmask_kernel<<<4096, 256, 0, stream>>>(wptr[i], sbuf + i, mptr[i], wn[i]);
    }

    rmsq_kernel<<<TOK, 256, 0, stream>>>(x, xq, invs);

    dim3 grid(TOK / 128, HID / 32);
    gemm1_kernel<<<grid, 256, 0, stream>>>(xq, invs, wgm, wum, p, pl0, pl1, pl2);

    gemm2_kernel<<<grid, 256, 0, stream>>>(pl0, pl1, pl2, wdm, invs, p, out);
}
